// Conv2d_78640851189797
// MI455X (gfx1250) — matrix operand order repaired and checked by reading, not compiled, not run on hardware
//
#include <hip/hip_runtime.h>

// ---------------------------------------------------------------------------
// 3x3 stride-1 pad-1 conv (NCHW), implicit GEMM on CDNA5 WMMA.
//
// Roofline (MI455X): 77.3 GFLOP, >=1.07 GB HBM -> 46 us floor @ 23.3 TB/s.
// AI ~72 FLOP/B. fp32 WMMA (16x16x4) would be compute-bound, so we run
// split-bf16 ("bf16x3"): x = xh+xl, w = wh+wl, accumulate
// xh*wh + xh*wl + xl*wh in the f32 accumulator of v_wmma_f32_16x16x32_bf16.
// ~16-17 mantissa bits => fp32-class accuracy at ~2.5-3x f32-WMMA speed.
//
// GEMM mapping: M=Cout(32), K=Cin*9 (looped as 9 positions x K=32), N=pixels.
// WG = 4 waves = 2 cout halves x 2 pixel pairs; wave owns two 16x16 tiles.
// LDS holds bf16 hi/lo weights [pos][cout][ci] and a 3x66x32 input halo tile
// [row][col][ci]; ci-contiguous so every WMMA fragment is two ds_load_b128.
//
// Fragment layouts (per CDNA5 ISA 7.12.2/7.12.4):
//   A 16x32 (16-bit): lanes 0-15: K {0..7}U{16..23}; lanes 16-31: {8..15}U{24..31}
//   B 32x16 (16-bit): ISA order as A; this code uses K 0..15 (lanes 0-15), K 16..31 (lanes 16-31) for A and B
//   D 16x16 (f32):    lanes 0-15: M=r (VGPR r);      lanes 16-31: M=r+8
// ---------------------------------------------------------------------------

typedef __bf16 bf16x8  __attribute__((ext_vector_type(8)));
typedef __bf16 bf16x16 __attribute__((ext_vector_type(16)));
typedef float  f32x8   __attribute__((ext_vector_type(8)));

#define IMG_H 512
#define IMG_W 512
#define CIN 32
#define COUT 32
#define NBATCH 16
#define TILE_W 64
#define XCOLS (TILE_W + 2)   // 66 columns incl. halo
#define NTHREADS 128         // 4 waves (wave32)

static __device__ __forceinline__ unsigned short f32_to_bf16_rne(float f) {
  unsigned int u = __float_as_uint(f);
  u += 0x7fffu + ((u >> 16) & 1u);        // round-to-nearest-even
  return (unsigned short)(u >> 16);
}
static __device__ __forceinline__ float bf16u_to_f32(unsigned short h) {
  return __uint_as_float(((unsigned int)h) << 16);
}

// Two 16B LDS reads at p and p+second_off_elems elements, packed into one fragment.
static __device__ __forceinline__ bf16x16 lds_frag(const unsigned short* p,
                                                   int second_off_elems) {
  bf16x8 lo = *(const bf16x8*)p;                        // ds_load_b128
  bf16x8 hi = *(const bf16x8*)(p + second_off_elems);   // ds_load_b128
  return __builtin_shufflevector(lo, hi, 0, 1, 2, 3, 4, 5, 6, 7,
                                 8, 9, 10, 11, 12, 13, 14, 15);
}

__global__ __launch_bounds__(NTHREADS) void conv3x3_wmma_bf16x3(
    const float* __restrict__ x, const float* __restrict__ wgt,
    float* __restrict__ out) {
  __shared__ __align__(16) unsigned short Wh[9 * COUT * CIN];
  __shared__ __align__(16) unsigned short Wl[9 * COUT * CIN];
  __shared__ __align__(16) unsigned short Xh[3 * XCOLS * CIN];
  __shared__ __align__(16) unsigned short Xl[3 * XCOLS * CIN];

  const int t  = threadIdx.x;
  const int w0 = blockIdx.x * TILE_W;
  const int h  = blockIdx.y;
  const int b  = blockIdx.z;

  // ---- Stage weights as bf16 hi/lo, layout [pos][cout][ci] ----
  for (int f = t; f < 9 * COUT * CIN; f += NTHREADS) {
    const int ci = f & 31, co = (f >> 5) & 31, pos = f >> 10;
    const float v = wgt[(co * CIN + ci) * 9 + pos];
    const unsigned short hs = f32_to_bf16_rne(v);
    Wh[f] = hs;
    Wl[f] = f32_to_bf16_rne(v - bf16u_to_f32(hs));
  }

  // ---- Stage input halo tile (rows h-1..h+1, cols w0-1..w0+TILE_W) as
  //      bf16 hi/lo, layout [row][col][ci]; zero-fill the pad border ----
  for (int f = t; f < 3 * XCOLS * CIN; f += NTHREADS) {
    const int col = f % XCOLS;
    const int rem = f / XCOLS;
    const int ci  = rem & 31;
    const int row = rem >> 5;
    const int hin = h + row - 1;
    const int win = w0 + col - 1;
    float v = 0.0f;
    if ((unsigned)hin < IMG_H && (unsigned)win < IMG_W)
      v = x[(((size_t)b * CIN + ci) * IMG_H + hin) * IMG_W + win];
    const unsigned short hs = f32_to_bf16_rne(v);
    const int idx = (row * XCOLS + col) * CIN + ci;
    Xh[idx] = hs;
    Xl[idx] = f32_to_bf16_rne(v - bf16u_to_f32(hs));
  }
  __syncthreads();

  // ---- Wave tiling: wid -> (cout half, pixel pair); two tiles per wave ----
  const int lane  = t & 31;
  const int wid   = t >> 5;
  const int co0   = (wid & 1) * 16;
  const int wp    = wid >> 1;          // pixels wp*32 .. wp*32+31
  const int m     = lane & 15;         // A row (cout) / B col (pixel) / D col
  const int khalf = lane >> 4;         // lane half selects K sub-pattern
  const int ka0   = khalf * 16;        // A: contiguous run {ka0..ka0+15}
  const int kb0   = khalf * 16;        // B: contiguous run {kb0..kb0+15}

  f32x8 acc[2] = {{}, {}};

#pragma unroll
  for (int pos = 0; pos < 9; ++pos) {
    const int kh = pos / 3, kw = pos % 3;
    // A fragment (weights): M = co0+m, contiguous K run (same order as A).
    const int ab = (pos * COUT + co0 + m) * CIN + ka0;
    const bf16x16 ah = lds_frag(Wh + ab, 8);
    const bf16x16 al = lds_frag(Wl + ab, 8);

#pragma unroll
    for (int half = 0; half < 2; ++half) {
      // B fragment (im2col): N = pixel p, contiguous K run (same order as A).
      const int p  = wp * 32 + half * 16 + m;
      const int bb = (kh * XCOLS + (p + kw)) * CIN + kb0;
      const bf16x16 bh = lds_frag(Xh + bb, 8);
      const bf16x16 bl = lds_frag(Xl + bb, 8);

      acc[half] = __builtin_amdgcn_wmma_f32_16x16x32_bf16(
          false, al, false, bh, (short)0, acc[half], false, false);
      acc[half] = __builtin_amdgcn_wmma_f32_16x16x32_bf16(
          false, ah, false, bl, (short)0, acc[half], false, false);
      acc[half] = __builtin_amdgcn_wmma_f32_16x16x32_bf16(
          false, ah, false, bh, (short)0, acc[half], false, false);
    }
  }

  // ---- Store: 16x16 f32 D layout: lanes 0-15 M=r, lanes 16-31 M=r+8 ----
  const int mrow0 = khalf * 8;
#pragma unroll
  for (int half = 0; half < 2; ++half) {
    const int p = wp * 32 + half * 16 + m;
    float* o = out +
        ((((size_t)b * COUT + co0 + mrow0) * IMG_H) + h) * IMG_W + (w0 + p);
#pragma unroll
    for (int r = 0; r < 8; ++r)
      o[(size_t)r * IMG_H * IMG_W] = acc[half][r];
  }
}

extern "C" void kernel_launch(void* const* d_in, const int* in_sizes, int n_in,
                              void* d_out, int out_size, void* d_ws, size_t ws_size,
                              hipStream_t stream) {
  (void)in_sizes; (void)n_in; (void)out_size; (void)d_ws; (void)ws_size;
  const float* x   = (const float*)d_in[0];
  const float* wgt = (const float*)d_in[1];
  float* out       = (float*)d_out;
  dim3 grid(IMG_W / TILE_W, IMG_H, NBATCH);
  conv3x3_wmma_bf16x3<<<grid, NTHREADS, 0, stream>>>(x, wgt, out);
}
